// GenNeuronStates_36146444763439
// MI455X (gfx1250) — matrix operand order repaired and checked by reading, not compiled, not run on hardware
//
#include <hip/hip_runtime.h>

#define BATCH 4
#define NSEQ  2048
#define HEADS 16
#define DQ    64
#define DV    64
#define TQ    128      // query rows per workgroup (8 waves x 16)
#define KT    64       // key rows per K/V tile
#define NTILE (NSEQ / KT)

typedef __bf16  v8bf  __attribute__((ext_vector_type(8)));
typedef __bf16  v16bf __attribute__((ext_vector_type(16)));
typedef float   v8f   __attribute__((ext_vector_type(8)));

typedef unsigned u32x4 __attribute__((ext_vector_type(4)));
typedef int      i32x4 __attribute__((ext_vector_type(4)));
typedef int      i32x8 __attribute__((ext_vector_type(8)));

__device__ __forceinline__ __bf16 f2bf(float f) {
  unsigned u = __builtin_bit_cast(unsigned, f);
  u += 0x7FFFu + ((u >> 16) & 1u);          // round-to-nearest-even
  unsigned short h = (unsigned short)(u >> 16);
  return __builtin_bit_cast(__bf16, h);
}

// Assemble a 32-byte WMMA operand from two 16-byte LDS chunks (-> 2x ds_load_b128).
__device__ __forceinline__ v16bf ld_frag(const __bf16* p0, const __bf16* p1) {
  v8bf lo = *(const v8bf*)p0;
  v8bf hi = *(const v8bf*)p1;
  v16bf f;
#pragma unroll
  for (int i = 0; i < 8; ++i) { f[i] = lo[i]; f[i + 8] = hi[i]; }
  return f;
}

__global__ __launch_bounds__(256)
void flash_attn_bf16(const float* __restrict__ Q,
                     const float* __restrict__ K,
                     const float* __restrict__ V,
                     const float* __restrict__ bias,
                     float* __restrict__ out)
{
  __shared__ __bf16 Qs[TQ * DQ];        // 16 KB  [row][q]
  __shared__ __bf16 Ks[KT * DQ];        //  8 KB  [m][q]
  __shared__ __bf16 VTs[DV * KT];       //  8 KB  [v][m]  (transposed)
  __shared__ __bf16 Ps[8 * 16 * KT];    // 16 KB  per-wave [row][m]
  __shared__ float  Bs[TQ * KT];        // 32 KB  bias tile [row][m], filled by TDM

  const int tid  = threadIdx.x;
  const int lane = tid & 31;
  const int w    = tid >> 5;
  const int half = lane >> 4;
  const int lr   = lane & 15;

  const int bh = blockIdx.x & (BATCH * HEADS - 1);
  const int qt = blockIdx.x / (BATCH * HEADS);
  const int b  = bh >> 4;
  const int h  = bh & 15;

  const float* qg = Q + (size_t)b * NSEQ * (DQ * HEADS);
  const float* kg = K + (size_t)b * NSEQ * (DQ * HEADS);
  const float* vg = V + (size_t)b * NSEQ * (DV * HEADS);

  const int q0 = qt * TQ;

  // ---- Stage this block's Q tile (TQ x DQ, head-deinterleaved) as bf16 ----
#pragma unroll
  for (int i = 0; i < (TQ * DQ) / 256; ++i) {
    int e = tid + i * 256;
    int r = e >> 6, q = e & 63;
    Qs[r * DQ + q] = f2bf(qg[(size_t)(q0 + r) * (DQ * HEADS) + q * HEADS + h]);
  }
  __syncthreads();

  // Persistent A-fragments of Q (two 32-wide k-chunks), per ISA A-layout.
  const int rbase = w * 16;
  v16bf qa[2];
#pragma unroll
  for (int qc = 0; qc < 2; ++qc) {
    const __bf16* p = Qs + (rbase + lr) * DQ + qc * 32 + half * 8;
    qa[qc] = ld_frag(p, p + 16);
  }

  // Online-softmax state; lane's row r maps to query row rbase + r + 8*half.
  float run_m[8], run_l[8];
  v8f o[4];
#pragma unroll
  for (int r = 0; r < 8; ++r) { run_m[r] = -1e30f; run_l[r] = 0.0f; }
#pragma unroll
  for (int vt = 0; vt < 4; ++vt)
#pragma unroll
    for (int r = 0; r < 8; ++r) o[vt][r] = 0.0f;

  const float scale = 0.125f;               // 1/sqrt(QLEN)
  const float L2E   = 1.44269504088896340736f;

  // TDM descriptor pieces that never change across k-tiles (2D tensor: TQ x KT
  // f32 tile out of the NSEQ x NSEQ bias matrix).
  const unsigned lds_bias = (unsigned)(uintptr_t)&Bs[0];

  for (int kt = 0; kt < NTILE; ++kt) {
    const int m0 = kt * KT;
    __syncthreads();                        // protect LDS tiles from prior readers

    // ---- Kick off TDM: bias[q0:q0+TQ, m0:m0+KT] f32 -> LDS (wave 0 only) ----
    if (w == 0) {
      unsigned long long ga =
          (unsigned long long)(uintptr_t)(bias + (size_t)q0 * NSEQ + m0);
      u32x4 g0;
      g0[0] = 1u;                                       // count=1, user D#
      g0[1] = lds_bias;                                 // lds_addr
      g0[2] = (unsigned)(ga & 0xFFFFFFFFu);             // global_addr[31:0]
      g0[3] = (unsigned)((ga >> 32) & 0x01FFFFFFu)      // global_addr[56:32]
              | (2u << 30);                             // type = 2 ("image")
      i32x8 g1;
      g1[0] = (2 << 16);                 // data_size = 4B; no mask/pad/iterate
      g1[1] = (int)((unsigned)NSEQ << 16);   // tensor_dim0[15:0] = 2048
      g1[2] = (int)((unsigned)NSEQ << 16);   // dim0 hi=0 | tensor_dim1[15:0]
      g1[3] = (int)((unsigned)KT << 16);     // dim1 hi=0 | tile_dim0 = 64
      g1[4] = TQ;                            // tile_dim1 = 128, tile_dim2 = 0
      g1[5] = NSEQ;                          // tensor_dim0_stride = 2048
      g1[6] = 0;                             // stride hi, tensor_dim1_stride lo
      g1[7] = 0;
      i32x4 gz4 = {0, 0, 0, 0};              // groups 2/3 unused (2D tensor)
      i32x8 gz8 = {0, 0, 0, 0, 0, 0, 0, 0};  // extra group (6-arg form), zeroed
      __builtin_amdgcn_tensor_load_to_lds(g0, g1, gz4, gz4, gz8, 0);
    }

    // ---- Cooperative K / V^T tile load (bf16), overlapped with the TDM ----
#pragma unroll
    for (int i = 0; i < (KT * DQ) / 256; ++i) {
      int e = tid + i * 256;
      int m = e >> 6, d = e & 63;
      const float* kp = kg + (size_t)(m0 + m) * (DQ * HEADS) + d * HEADS + h;
      const float* vp = vg + (size_t)(m0 + m) * (DV * HEADS) + d * HEADS + h;
      Ks[m * DQ + d]  = f2bf(*kp);
      VTs[d * KT + m] = f2bf(*vp);
      if (kt + 1 < NTILE) {
        __builtin_prefetch(kp + KT * (DQ * HEADS), 0, 0);
        __builtin_prefetch(vp + KT * (DV * HEADS), 0, 0);
      }
    }
    if (w == 0) __builtin_amdgcn_s_wait_tensorcnt(0);
    __syncthreads();

    // ---- S = (Q K^T)/8 - bias : four 16x16 column tiles ----
    float xs[4][8];
#pragma unroll
    for (int mj = 0; mj < 4; ++mj) {
      v8f s;
#pragma unroll
      for (int r = 0; r < 8; ++r) s[r] = 0.0f;
#pragma unroll
      for (int qc = 0; qc < 2; ++qc) {
        const __bf16* kb = Ks + (mj * 16 + lr) * DQ + qc * 32 + half * 8;
        v16bf bf = ld_frag(kb, kb + 16);    // B-layout: lane=col m, q = 8h+s / 16+8h+s-8
        s = __builtin_amdgcn_wmma_f32_16x16x32_bf16(
                false, qa[qc], false, bf, (short)0, s, false, false);
      }
#pragma unroll
      for (int r = 0; r < 8; ++r) {
        int rloc = rbase + r + 8 * half;    // row within the TQ tile
        xs[mj][r] = s[r] * scale - Bs[rloc * KT + mj * 16 + lr];
      }
    }

    // ---- Online softmax over this key tile ----
#pragma unroll
    for (int r = 0; r < 8; ++r) {
      float tm = xs[0][r];
#pragma unroll
      for (int mj = 1; mj < 4; ++mj) tm = fmaxf(tm, xs[mj][r]);
#pragma unroll
      for (int msk = 8; msk >= 1; msk >>= 1)
        tm = fmaxf(tm, __shfl_xor(tm, msk, 32));
      float nm   = fmaxf(run_m[r], tm);
      float corr = exp2f((run_m[r] - nm) * L2E);
      float ts   = 0.0f;
#pragma unroll
      for (int mj = 0; mj < 4; ++mj) {
        float p = exp2f((xs[mj][r] - nm) * L2E);
        xs[mj][r] = p;
        ts += p;
      }
#pragma unroll
      for (int msk = 8; msk >= 1; msk >>= 1) ts += __shfl_xor(ts, msk, 32);
      run_l[r] = run_l[r] * corr + ts;
      run_m[r] = nm;
#pragma unroll
      for (int vt = 0; vt < 4; ++vt) o[vt][r] *= corr;
    }

    // ---- Spill P (16 x KT) to this wave's LDS strip, reload as A-operand ----
    __bf16* Pw = Ps + w * (16 * KT);
#pragma unroll
    for (int mj = 0; mj < 4; ++mj)
#pragma unroll
      for (int r = 0; r < 8; ++r)
        Pw[(r + 8 * half) * KT + mj * 16 + lr] = f2bf(xs[mj][r]);

    // ---- O += P V : A = P rows, B = V^T columns ----
#pragma unroll
    for (int ac = 0; ac < 2; ++ac) {
      const __bf16* pb = Pw + lr * KT + ac * 32 + half * 8;
      v16bf pa = ld_frag(pb, pb + 16);
#pragma unroll
      for (int vt = 0; vt < 4; ++vt) {
        const __bf16* vb = VTs + (vt * 16 + lr) * KT + ac * 32 + half * 8;
        v16bf vf = ld_frag(vb, vb + 16);
        o[vt] = __builtin_amdgcn_wmma_f32_16x16x32_bf16(
                    false, pa, false, vf, (short)0, o[vt], false, false);
      }
    }
  }

  // ---- Epilogue: normalize and store in [B,H,N,V]-contiguous order ----
#pragma unroll
  for (int r = 0; r < 8; ++r) {
    float inv = 1.0f / run_l[r];
    int nrow  = q0 + rbase + r + 8 * half;
    size_t ob = (((size_t)(b * HEADS + h)) * NSEQ + nrow) * DV;
#pragma unroll
    for (int vt = 0; vt < 4; ++vt)
      out[ob + vt * 16 + lr] = o[vt][r] * inv;
  }
}

extern "C" void kernel_launch(void* const* d_in, const int* in_sizes, int n_in,
                              void* d_out, int out_size, void* d_ws, size_t ws_size,
                              hipStream_t stream) {
  (void)in_sizes; (void)n_in; (void)out_size; (void)d_ws; (void)ws_size;
  const float* Q    = (const float*)d_in[0];
  const float* K    = (const float*)d_in[1];
  const float* V    = (const float*)d_in[2];
  const float* bias = (const float*)d_in[3];
  float* out = (float*)d_out;

  dim3 grid((NSEQ / TQ) * BATCH * HEADS);   // 16 q-tiles * 64 (b,h) = 1024 WGs
  flash_attn_bf16<<<grid, 256, 0, stream>>>(Q, K, V, bias, out);
}
